// Fusion_24876450578934
// MI455X (gfx1250) — matrix operand order repaired and checked by reading, not compiled, not run on hardware
//
#include <hip/hip_runtime.h>

#define Bsz 2
#define CC  64
#define DI  128
#define NN  16
#define LL  4096
#define BL  (Bsz * LL)
#define CHUNK 64
#define NCHUNK (LL / CHUNK)   // 64

typedef __bf16 bf16;
typedef bf16  v16bf __attribute__((ext_vector_type(16)));
typedef float v8f   __attribute__((ext_vector_type(8)));
typedef unsigned int u32x4 __attribute__((ext_vector_type(4)));
typedef int i32x4 __attribute__((ext_vector_type(4)));
typedef int i32x8 __attribute__((ext_vector_type(8)));

#if __has_builtin(__builtin_amdgcn_tensor_load_to_lds) && \
    __has_builtin(__builtin_amdgcn_s_wait_tensorcnt)
#define USE_TDM 1
#else
#define USE_TDM 0
#endif

// ---- WMMA fragment helpers (16x16x32 bf16, wave32) -------------------------
// A (16x32, row-per-lane): lane m=lane&15 holds row m; K halves split on lane>=16.
__device__ __forceinline__ v16bf a_frag(const bf16* row, int k0, int lane) {
  const int base = (lane & 16) ? 16 : 0;
  v16bf a;
#pragma unroll
  for (int e = 0; e < 16; ++e) {
    a[e]     = row[k0 + base + e];
  }
  return a;
}
// B (32x16, column-per-lane): lane n holds B[kb..kb+15][n] = Wrow[kb..kb+15]
__device__ __forceinline__ v16bf b_frag(const float* wrow, int k0, int lane) {
  const int kb = k0 + ((lane & 16) ? 16 : 0);
  v16bf b;
#pragma unroll
  for (int e = 0; e < 16; ++e) b[e] = (bf16)wrow[kb + e];
  return b;
}
__device__ __forceinline__ v8f wmma_bf16(v16bf a, v16bf b, v8f c) {
  return __builtin_amdgcn_wmma_f32_16x16x32_bf16(false, a, false, b, (short)0, c,
                                                 false, false);
}
__device__ __forceinline__ float silu_f(float v) { return v / (1.f + __expf(-v)); }

// ---- K0: residual init out = x + y + z -------------------------------------
__global__ void add3_kernel(const float* __restrict__ a, const float* __restrict__ b,
                            const float* __restrict__ c, float* __restrict__ o, int n) {
  int i = blockIdx.x * blockDim.x + threadIdx.x;
  if (i < n) o[i] = a[i] + b[i] + c[i];
}

// ---- K1: embed1x1 + rmsnorm + input projection (WMMA) ----------------------
__global__ void embed_rms_proj_kernel(const float* __restrict__ in,
                                      const float* __restrict__ pe_w,
                                      const float* __restrict__ pe_b,
                                      const float* __restrict__ rms_w,
                                      const float* __restrict__ in_w,
                                      float* __restrict__ out, int Dtot) {
  __shared__ bf16  xt[16 * 72];
  __shared__ float xe[16 * 68];
  __shared__ float rstd[16];
  const int tile = blockIdx.x;          // BL/16 tiles
  const int b = (tile * 16) / LL, l0 = (tile * 16) % LL;
  const int t = threadIdx.x, lane = t & 31, wave = t >> 5;

  // warm GL2 with the projection weights this block will sweep (global_prefetch_b8)
  for (int r = t; r < Dtot; r += 128) __builtin_prefetch(in_w + (size_t)r * CC, 0, 1);

  for (int e = t; e < 16 * CC; e += 128) {
    int c = e >> 4, p = e & 15;
    xt[p * 72 + c] = (bf16)in[(size_t)(b * CC + c) * LL + l0 + p];
  }
  __syncthreads();

  {  // GEMM1: xe_raw = X @ pe_w^T
    const int n0 = wave * 16, m = lane & 15, n = lane & 15;
    const int mbase = (lane & 16) ? 8 : 0;
    v8f acc = {};
#pragma unroll
    for (int k0 = 0; k0 < CC; k0 += 32)
      acc = wmma_bf16(a_frag(&xt[m * 72], k0, lane),
                      b_frag(&pe_w[(n0 + n) * CC], k0, lane), acc);
    const float bias = pe_b[n0 + n];
#pragma unroll
    for (int r = 0; r < 8; ++r) xe[(mbase + r) * 68 + n0 + n] = acc[r] + bias;
  }
  __syncthreads();

  if (t < 16) {
    float ss = 0.f;
    for (int c = 0; c < CC; ++c) { float v = xe[t * 68 + c]; ss += v * v; }
    rstd[t] = rsqrtf(ss * (1.0f / CC) + 1e-5f);
  }
  __syncthreads();

  for (int e = t; e < 16 * CC; e += 128) {
    int c = e & 63, p = e >> 6;
    xt[p * 72 + c] = (bf16)(xe[p * 68 + c] * rstd[p] * rms_w[c]);
  }
  __syncthreads();

  const int ntiles = Dtot >> 4;  // GEMM2: proj = xe @ in_w^T
  for (int nt = wave; nt < ntiles; nt += 4) {
    const int d0 = nt * 16, m = lane & 15, n = lane & 15;
    const int mbase = (lane & 16) ? 8 : 0;
    v8f acc = {};
#pragma unroll
    for (int k0 = 0; k0 < CC; k0 += 32)
      acc = wmma_bf16(a_frag(&xt[m * 72], k0, lane),
                      b_frag(&in_w[(size_t)(d0 + n) * CC], k0, lane), acc);
    float* op = out + (size_t)(b * Dtot + d0 + n) * LL + l0 + mbase;
#pragma unroll
    for (int r = 0; r < 8; ++r) op[r] = acc[r];
  }
}

// ---- K2: causal depthwise conv (K=3) + SiLU, all 3 streams in one launch ---
__global__ void dwconv3_kernel(const float* __restrict__ xz,
                               const float* __restrict__ ir,
                               const float* __restrict__ mr,
                               const float* __restrict__ wx, const float* __restrict__ bx,
                               const float* __restrict__ wi, const float* __restrict__ bi,
                               const float* __restrict__ wm, const float* __restrict__ bm,
                               float* __restrict__ ox, float* __restrict__ oi,
                               float* __restrict__ om) {
  const int s = blockIdx.y;
  const float* in; size_t bstride; const float* w; const float* bias; float* out;
  if (s == 0)      { in = xz; bstride = (size_t)2 * DI * LL; w = wx; bias = bx; out = ox; }
  else if (s == 1) { in = ir; bstride = (size_t)DI * LL;     w = wi; bias = bi; out = oi; }
  else             { in = mr; bstride = (size_t)DI * LL;     w = wm; bias = bm; out = om; }
  const int idx = blockIdx.x * blockDim.x + threadIdx.x;
  const int l = idx & (LL - 1);
  const int d = (idx >> 12) & (DI - 1);
  const int b = idx / (DI * LL);
  const float* xp = in + (size_t)b * bstride + (size_t)d * LL;
  float v = w[d * 3 + 2] * xp[l];
  if (l >= 1) v += w[d * 3 + 1] * xp[l - 1];
  if (l >= 2) v += w[d * 3 + 0] * xp[l - 2];
  v += bias[d];
  out[(size_t)(b * DI + d) * LL + l] = silu_f(v);
}

// ---- K3: delta / Bm / Cm projections (WMMA) + softplus expansion ------------
__global__ void proj_small_kernel(const float* __restrict__ xc,
                                  const float* __restrict__ info,
                                  const float* __restrict__ mk,
                                  const float* __restrict__ Bw,
                                  const float* __restrict__ Cw,
                                  const float* __restrict__ dw,
                                  const float* __restrict__ dtw,
                                  const float* __restrict__ dtb,
                                  float* __restrict__ Bm, float* __restrict__ Cm,
                                  float* __restrict__ delta) {
  __shared__ bf16  tx[16 * 136], ti[16 * 136], tm[16 * 136];
  __shared__ float dr[16 * 20];
  const int tile = blockIdx.x;
  const int b = (tile * 16) / LL, l0 = (tile * 16) % LL;
  const int t = threadIdx.x, lane = t & 31, wave = t >> 5;

  for (int e = t; e < 16 * DI; e += 128) {
    int d = e >> 4, m = e & 15;
    size_t gi = (size_t)(b * DI + d) * LL + l0 + m;
    tx[m * 136 + d] = (bf16)xc[gi];
    ti[m * 136 + d] = (bf16)info[gi];
    tm[m * 136 + d] = (bf16)mk[gi];
  }
  __syncthreads();

  const int m = lane & 15, n = lane & 15;
  const int mbase = (lane & 16) ? 8 : 0;
  if (wave == 0) {  // Cm = xc . Cw
    v8f acc = {};
#pragma unroll
    for (int k0 = 0; k0 < DI; k0 += 32)
      acc = wmma_bf16(a_frag(&tx[m * 136], k0, lane), b_frag(&Cw[n * DI], k0, lane), acc);
    float* op = Cm + (size_t)(b * NN + n) * LL + l0 + mbase;
#pragma unroll
    for (int r = 0; r < 8; ++r) op[r] = acc[r];
  } else if (wave == 1) {  // Bm = info . Bw
    v8f acc = {};
#pragma unroll
    for (int k0 = 0; k0 < DI; k0 += 32)
      acc = wmma_bf16(a_frag(&ti[m * 136], k0, lane), b_frag(&Bw[n * DI], k0, lane), acc);
    float* op = Bm + (size_t)(b * NN + n) * LL + l0 + mbase;
#pragma unroll
    for (int r = 0; r < 8; ++r) op[r] = acc[r];
  } else if (wave == 2) {  // delta_r = mk . delta_w (4 valid N cols)
    v8f acc = {};
#pragma unroll
    for (int k0 = 0; k0 < DI; k0 += 32) {
      const int kb = k0 + ((lane & 16) ? 16 : 0);
      v16bf bf;
#pragma unroll
      for (int e = 0; e < 16; ++e) bf[e] = (n < 4) ? (bf16)dw[n * DI + kb + e] : (bf16)0.0f;
      acc = wmma_bf16(a_frag(&tm[m * 136], k0, lane), bf, acc);
    }
#pragma unroll
    for (int r = 0; r < 8; ++r) dr[(mbase + r) * 20 + n] = acc[r];
  }
  __syncthreads();

  for (int e = t; e < 16 * DI; e += 128) {
    int d = e & (DI - 1), ml = e >> 7;
    float s = dtb[d];
#pragma unroll
    for (int r = 0; r < 4; ++r) s += dr[ml * 20 + r] * dtw[d * 4 + r];
    float sp = (s > 20.f) ? s : log1pf(__expf(s));
    delta[(size_t)(b * DI + d) * LL + l0 + ml] = sp;
  }
}

// ---- K4a: per-chunk scan transfer ------------------------------------------
__global__ void scanA_kernel(const float* __restrict__ u,
                             const float* __restrict__ delta,
                             const float* __restrict__ Bm,
                             const float* __restrict__ Alog,
                             float* __restrict__ Pc, float* __restrict__ Qc) {
  const int blk = blockIdx.x;
  const int dg = blk & 7, b = (blk >> 3) & 1, c = blk >> 4;
  const int t = threadIdx.x;
  const int n = t & 15, d = dg * 16 + (t >> 4);
  const float A = -__expf(Alog[d * NN + n]);
  const int l0 = c * CHUNK;
  const float* dp = delta + (size_t)(b * DI + d) * LL + l0;
  const float* up = u + (size_t)(b * DI + d) * LL + l0;
  const float* Bp = Bm + (size_t)(b * NN + n) * LL + l0;
  float S = 0.f, Q = 0.f;
  for (int l = 0; l < CHUNK; ++l) {
    const float dv = dp[l];
    const float dA = __expf(dv * A);
    Q = fmaf(dA, Q, dv * Bp[l] * up[l]);
    S += dv;
  }
  const size_t idx = ((size_t)((c * Bsz + b) * DI + d)) * NN + n;
  Pc[idx] = __expf(A * S);
  Qc[idx] = Q;
}

// ---- K4b: serial scan over the 64 chunk carries ----------------------------
__global__ void scanB_kernel(const float* __restrict__ Pc,
                             const float* __restrict__ Qc,
                             float* __restrict__ Hc) {
  const int t = blockIdx.x * blockDim.x + threadIdx.x;  // Bsz*DI*NN
  const int n = t & 15, d = (t >> 4) & (DI - 1), b = t >> 11;
  float h = 0.f;
  for (int c = 0; c < NCHUNK; ++c) {
    const size_t idx = ((size_t)((c * Bsz + b) * DI + d)) * NN + n;
    Hc[idx] = h;
    h = fmaf(Pc[idx], h, Qc[idx]);
  }
}

// ---- K4c: replay chunks from carry-in, emit gated y ------------------------
__global__ void scanC_kernel(const float* __restrict__ u,
                             const float* __restrict__ delta,
                             const float* __restrict__ Bm,
                             const float* __restrict__ Cm,
                             const float* __restrict__ Alog,
                             const float* __restrict__ xz,
                             const float* __restrict__ Hc,
                             float* __restrict__ y) {
  const int blk = blockIdx.x;
  const int dg = blk & 7, b = (blk >> 3) & 1, c = blk >> 4;
  const int t = threadIdx.x;
  const int n = t & 15, d = dg * 16 + (t >> 4);
  const float A = -__expf(Alog[d * NN + n]);
  const int l0 = c * CHUNK;
  const float* dp = delta + (size_t)(b * DI + d) * LL + l0;
  const float* up = u + (size_t)(b * DI + d) * LL + l0;
  const float* Bp = Bm + (size_t)(b * NN + n) * LL + l0;
  const float* Cp = Cm + (size_t)(b * NN + n) * LL + l0;
  const float* xa = xz + (size_t)(b * 2 * DI + DI + d) * LL + l0;
  float* yp = y + (size_t)(b * DI + d) * LL + l0;
  float h = Hc[((size_t)((c * Bsz + b) * DI + d)) * NN + n];
  for (int l = 0; l < CHUNK; ++l) {
    const float dv = dp[l];
    const float dA = __expf(dv * A);
    h = fmaf(dA, h, dv * Bp[l] * up[l]);
    float cv = h * Cp[l];
    cv += __shfl_xor(cv, 1, 32);
    cv += __shfl_xor(cv, 2, 32);
    cv += __shfl_xor(cv, 4, 32);
    cv += __shfl_xor(cv, 8, 32);
    if (n == 0) yp[l] = cv * silu_f(xa[l]);
  }
}

// ---- K5: output projection (WMMA); tile staged by the Tensor Data Mover ----
// Gathers a 16(l) x 128(d) tile from [d][l]-major memory: 128 strided rows of
// 64B each == a 2D D# (tile_dim0=16, tile_dim1=128, tensor_dim0_stride=L).
__global__ void outproj_kernel(const float* __restrict__ y,   // [B][128][L]
                               const float* __restrict__ ow,  // [64][128]
                               float* __restrict__ out) {     // [B][64][L] +=
  __shared__ float tyf[16 * DI];   // TDM destination, packed [d][m]
  __shared__ bf16  ty[16 * 136];
  const int tile = blockIdx.x;
  const int b = (tile * 16) / LL, l0 = (tile * 16) % LL;
  const int t = threadIdx.x, lane = t & 31, wave = t >> 5;
  for (int r = t; r < CC; r += 128) __builtin_prefetch(ow + (size_t)r * DI, 0, 1);

#if USE_TDM
  if (t < 32) {  // wave 0 issues one TDM gather of the whole tile
    const unsigned long long ga =
        (unsigned long long)(size_t)(y + (size_t)b * DI * LL + l0);
    const unsigned int lds = (unsigned int)(size_t)&tyf[0];
    u32x4 g0;
    g0[0] = 1u;                                        // count=1, user D#
    g0[1] = lds;                                       // lds_addr
    g0[2] = (unsigned int)(ga & 0xFFFFFFFFull);        // global_addr[31:0]
    g0[3] = (unsigned int)((ga >> 32) & 0x01FFFFFFull) // global_addr[56:32]
            | (2u << 30);                              // type = 2 (image)
    const unsigned long long td0 = (unsigned long long)LL;   // tensor_dim0
    const unsigned long long td1 = (unsigned long long)DI;   // tensor_dim1
    const unsigned long long st0 = (unsigned long long)LL;   // dim0 stride
    unsigned long long q0 = (2ull << 16)               // data_size = 4B
                          | ((td0 & 0xFFFFull) << 48);
    unsigned long long q1 = (td0 >> 16)                // tensor_dim0[31:16]
                          | (td1 << 16)                // tensor_dim1
                          | (16ull << 48);             // tile_dim0 = 16
    unsigned long long q2 = (unsigned long long)DI     // tile_dim1 = 128
                          | ((st0 & 0xFFFFFFFFull) << 32);
    unsigned long long q3 = (st0 >> 32) & 0xFFFFull;   // stride0[47:32]
    i32x8 g1;
    g1[0] = (int)(q0 & 0xFFFFFFFFull); g1[1] = (int)(q0 >> 32);
    g1[2] = (int)(q1 & 0xFFFFFFFFull); g1[3] = (int)(q1 >> 32);
    g1[4] = (int)(q2 & 0xFFFFFFFFull); g1[5] = (int)(q2 >> 32);
    g1[6] = (int)(q3 & 0xFFFFFFFFull); g1[7] = (int)(q3 >> 32);
    i32x4 z4 = {};
#if __clang_major__ >= 23
    i32x8 z8 = {};
    __builtin_amdgcn_tensor_load_to_lds(g0, g1, z4, z4, z8, 0);
#else
    __builtin_amdgcn_tensor_load_to_lds(g0, g1, z4, z4, 0);
#endif
    __builtin_amdgcn_s_wait_tensorcnt(0);
  }
  __syncthreads();
  for (int e = t; e < 16 * DI; e += 128) {   // f32 -> bf16 fragment tile
    int d = e >> 4, m = e & 15;
    ty[m * 136 + d] = (bf16)tyf[d * 16 + m];
  }
#else
  for (int e = t; e < 16 * DI; e += 128) {
    int d = e >> 4, m = e & 15;
    ty[m * 136 + d] = (bf16)y[(size_t)(b * DI + d) * LL + l0 + m];
  }
#endif
  __syncthreads();

  const int m = lane & 15, n = lane & 15;
  const int mbase = (lane & 16) ? 8 : 0;
  const int c0 = wave * 16;
  v8f acc = {};
#pragma unroll
  for (int k0 = 0; k0 < DI; k0 += 32)
    acc = wmma_bf16(a_frag(&ty[m * 136], k0, lane),
                    b_frag(&ow[(c0 + n) * DI], k0, lane), acc);
  float* op = out + (size_t)(b * CC + c0 + n) * LL + l0 + mbase;
#pragma unroll
  for (int r = 0; r < 8; ++r) op[r] += acc[r];
}

// ---- host orchestration -----------------------------------------------------
extern "C" void kernel_launch(void* const* d_in, const int* in_sizes, int n_in,
                              void* d_out, int out_size, void* d_ws, size_t ws_size,
                              hipStream_t stream) {
  const float* up   = (const float*)d_in[0];
  const float* mask = (const float*)d_in[1];
  const float* feats[3] = {(const float*)d_in[2], (const float*)d_in[3],
                           (const float*)d_in[4]};
  auto P = [&](int i) { return (const float*)d_in[i]; };

  float* ws = (float*)d_ws;
  const size_t F = (size_t)Bsz * DI * LL;
  const size_t BNL = (size_t)Bsz * NN * LL;
  const size_t CK = (size_t)NCHUNK * Bsz * DI * NN;
  float* xz       = ws;
  float* info_raw = ws + 2 * F;
  float* mk_raw   = ws + 3 * F;
  float* xc_c     = ws + 4 * F;
  float* info_c   = ws + 5 * F;
  float* mk_c     = ws + 6 * F;
  float* dlt      = ws + 7 * F;
  float* yb       = ws + 8 * F;
  float* BmW      = ws + 9 * F;
  float* CmW      = BmW + BNL;
  float* Pc       = CmW + BNL;
  float* Qc       = Pc + CK;
  float* Hc       = Qc + CK;

  const int total = Bsz * CC * LL;
  add3_kernel<<<(total + 255) / 256, 256, 0, stream>>>(feats[0], feats[1], feats[2],
                                                       (float*)d_out, total);

  const int ntile = BL / 16;
  const int sgrid = NCHUNK * Bsz * (DI / 16);
  for (int br = 0; br < 3; ++br) {
    const float* pe_x_w   = P(5)  + br * CC * CC;
    const float* pe_x_b   = P(6)  + br * CC;
    const float* rms_x_w  = P(7)  + br * CC;
    const float* conv_x_w = P(8)  + br * DI * 3;
    const float* conv_x_b = P(9)  + br * DI;
    const float* pe_i_w   = P(10) + br * CC * CC;
    const float* pe_i_b   = P(11) + br * CC;
    const float* rms_i_w  = P(12) + br * CC;
    const float* conv_i_w = P(13) + br * DI * 3;
    const float* conv_i_b = P(14) + br * DI;
    const float* pe_m_w   = P(15) + br * CC * CC;
    const float* pe_m_b   = P(16) + br * CC;
    const float* rms_m_w  = P(17) + br * CC;
    const float* conv_m_w = P(18) + br * DI * 3;
    const float* conv_m_b = P(19) + br * DI;
    const float* in_x_w   = P(20) + br * 2 * DI * CC;
    const float* in_i_w   = P(21) + br * DI * CC;
    const float* in_m_w   = P(22) + br * DI * CC;
    const float* delta_w  = P(23) + br * 4 * DI;
    const float* dt_w     = P(24) + br * DI * 4;
    const float* dt_b     = P(25) + br * DI;
    const float* B_w      = P(26) + br * NN * DI;
    const float* C_w      = P(27) + br * NN * DI;
    const float* A_log    = P(28) + br * DI * NN;
    const float* out_w    = P(29) + br * CC * DI;

    embed_rms_proj_kernel<<<ntile, 128, 0, stream>>>(feats[br], pe_x_w, pe_x_b,
                                                     rms_x_w, in_x_w, xz, 2 * DI);
    embed_rms_proj_kernel<<<ntile, 128, 0, stream>>>(up, pe_i_w, pe_i_b, rms_i_w,
                                                     in_i_w, info_raw, DI);
    embed_rms_proj_kernel<<<ntile, 128, 0, stream>>>(mask, pe_m_w, pe_m_b, rms_m_w,
                                                     in_m_w, mk_raw, DI);

    dim3 cgrid((Bsz * DI * LL) / 256, 3);
    dwconv3_kernel<<<cgrid, 256, 0, stream>>>(xz, info_raw, mk_raw,
                                              conv_x_w, conv_x_b, conv_i_w, conv_i_b,
                                              conv_m_w, conv_m_b, xc_c, info_c, mk_c);

    proj_small_kernel<<<ntile, 128, 0, stream>>>(xc_c, info_c, mk_c, B_w, C_w,
                                                 delta_w, dt_w, dt_b, BmW, CmW, dlt);

    scanA_kernel<<<sgrid, 256, 0, stream>>>(xc_c, dlt, BmW, A_log, Pc, Qc);
    scanB_kernel<<<(Bsz * DI * NN) / 256, 256, 0, stream>>>(Pc, Qc, Hc);
    scanC_kernel<<<sgrid, 256, 0, stream>>>(xc_c, dlt, BmW, CmW, A_log, xz, Hc, yb);

    outproj_kernel<<<ntile, 128, 0, stream>>>(yb, out_w, (float*)d_out);
  }
}
